// VGGplus1_66254165508305
// MI455X (gfx1250) — matrix operand order repaired and checked by reading, compiled, not run on hardware
//
#include <hip/hip_runtime.h>
#include <math.h>

// ---------------------------------------------------------------------------
// MI455X / gfx1250 implementation of the SparseVggNet-like reference.
// Activations: f16 NHWC. Convs: implicit-GEMM, one wave = 32(pixels)x16(ch)
// tile = two v_wmma_f32_16x16x32_f16 per K-step sharing one B fragment.
// im2col K->offset decode hoisted into an LDS table (no div/mod in hot loop).
// Weights pre-packed into WMMA B-fragment blocks. BN = LDS-atomic partials +
// global f32 atomics + fused normalize/scale/ReLU. 5x5 conv + Linear share a
// WMMA GEMM with global_prefetch on the streaming A operand.
// ---------------------------------------------------------------------------

typedef _Float16 half_t;
typedef __attribute__((ext_vector_type(16))) _Float16 v16h;
typedef __attribute__((ext_vector_type(8)))  _Float16 v8h;
typedef __attribute__((ext_vector_type(8)))  float    v8f;

#define WAVES_PER_WG 4
#define MAX_PATCH 6272   // >= KH * ((32-1)*stride + KW) * Cin  (3*65*32 = 6240)
#define MAX_KTAB  320    // >= ceil(Cin*KH*KW/32)*32            (9*32 = 288)

// -------------------------------- utility ----------------------------------

__global__ void k_zero(float* p, int n) {
  int i = blockIdx.x * blockDim.x + threadIdx.x;
  if (i < n) p[i] = 0.f;
}

__global__ void k_f32_to_f16(const float* __restrict__ in, half_t* __restrict__ out, long n) {
  long i = (long)blockIdx.x * blockDim.x + threadIdx.x;
  if (i < n) out[i] = (half_t)in[i];
}

// Pack OIHW f32 weights into WMMA B-fragment blocks.
// K index k = (kh*KW + kw)*Cin + ci  (matches NHWC im2col order).
// Block (kb, nb) = 512 halves, lane L holds column N = nb*16+(L&15), halves e =
// K = kb*32 + (L>=16 ? 8 : 0) + (e<8 ? e : e+8)  (ISA 7.12.2 B layout, wave32).
__global__ void k_pack(const float* __restrict__ w, half_t* __restrict__ pb,
                       int Cout, int Cin, int KH, int KW) {
  int Ktot = Cin * KH * KW;
  int nKB = (Ktot + 31) >> 5, nNB = (Cout + 15) >> 4;
  long total = (long)nKB * nNB * 512;
  long i = (long)blockIdx.x * blockDim.x + threadIdx.x;
  if (i >= total) return;
  int e = (int)(i & 15);
  int L = (int)((i >> 4) & 31);
  long blk = i >> 9;
  int nb = (int)(blk % nNB);
  int kb = (int)(blk / nNB);
  int k = kb * 32 + ((L & 16) ? 8 : 0) + (e < 8 ? e : e + 8);
  int o = nb * 16 + (L & 15);
  half_t v = (half_t)0.f;
  if (k < Ktot && o < Cout) {
    int ci = k % Cin;
    int r = k / Cin;
    int kw = r % KW;
    int kh = r / KW;
    v = (half_t)w[(((long)o * Cin + ci) * KH + kh) * KW + kw];
  }
  pb[i] = v;
}

// ------------------------- implicit-GEMM 3x3 conv --------------------------
// One wave = one 32(pixels)x16(channels) output tile: two 16x16 WMMA
// accumulators sharing one B fragment per K-step. The 32 pixels are
// consecutive output-x positions of one (n, oh) row; their overlapping input
// patch (KH x PW x Cin halves) is staged once in LDS. The K -> patch-offset
// decode is precomputed in a per-workgroup LDS table so the hot loop is just
// ds_load_u16 + add (no integer div/mod).
__global__ __launch_bounds__(128) void
k_conv_wmma(const half_t* __restrict__ in, const half_t* __restrict__ pb,
            half_t* __restrict__ out,
            int N, int H, int W, int Cin, int Cout, int KH, int KW,
            int stride, int pad, int OH, int OW, int Ctot, int coff) {
  __shared__ __align__(16) half_t patch[WAVES_PER_WG][MAX_PATCH];
  __shared__ unsigned short ktab[MAX_KTAB];

  const int wid  = threadIdx.x >> 5;
  const int lane = threadIdx.x & 31;

  const int nOwT = (OW + 31) >> 5;
  const int nCoT = (Cout + 15) >> 4;
  const int Ktot = Cin * KH * KW;
  const int nKB  = (Ktot + 31) >> 5;
  const int PW   = 31 * stride + KW;             // patch width in pixels

  // K -> patch base-offset table (identical for all tiles of this launch).
  for (int k = threadIdx.x; k < (nKB << 5); k += 128) {
    unsigned short v = 0xFFFFu;
    if (k < Ktot) {
      int ci = k % Cin;
      int r  = k / Cin;
      int kw = r % KW;
      int kh = r / KW;
      v = (unsigned short)((kh * PW + kw) * Cin + ci);
    }
    ktab[k] = v;
  }

  const long tiles = (long)N * OH * nOwT * nCoT;
  long t = (long)blockIdx.x * WAVES_PER_WG + wid;
  const bool valid = (t < tiles);
  long tt = valid ? t : 0;
  const int cot = (int)(tt % nCoT); tt /= nCoT;
  const int owt = (int)(tt % nOwT); tt /= nOwT;
  const int oh  = (int)(tt % OH);
  const int n   = (int)(tt / OH);
  const int ow0 = owt << 5;

  // Cooperative patch fill (zero pad out-of-bounds).
  const int patchElems = KH * PW * Cin;
  const int ih0 = oh * stride - pad;
  const int ix0 = ow0 * stride - pad;
  for (int i = lane; i < patchElems; i += 32) {
    int ci = i % Cin;
    int r  = i / Cin;
    int px = r % PW;
    int kh = r / PW;
    int ih = ih0 + kh;
    int ix = ix0 + px;
    half_t v = (half_t)0.f;
    if (ih >= 0 && ih < H && ix >= 0 && ix < W)
      v = in[(((long)n * H + ih) * W + ix) * Cin + ci];
    patch[wid][i] = v;
  }
  __syncthreads();

  const int m = lane & 15;                       // A-fragment row = pixel idx
  const int mOff0 = m * stride * Cin;
  const int mOff1 = mOff0 + 16 * stride * Cin;
  v8f acc0 = {}, acc1 = {};

  for (int kb = 0; kb < nKB; ++kb) {
    // A fragments (ISA 16-bit A 16x32): lane<16 -> K 0..7,16..23 ; +16 -> +8.
    v16h a0, a1;
#pragma unroll
    for (int e = 0; e < 16; ++e) {
      int koff = ((lane & 16) ? 8 : 0) + (e < 8 ? e : e + 8);
      unsigned short ba = ktab[(kb << 5) + koff];
      half_t v0 = (half_t)0.f, v1 = (half_t)0.f;
      if (ba != 0xFFFFu) {
        v0 = patch[wid][ba + mOff0];
        v1 = patch[wid][ba + mOff1];
      }
      a0[e] = v0;
      a1[e] = v1;
    }
    // B fragment: lane = column N (lane&15) of the packed 32x16 block, 16 contiguous K halves.
    const v16h* bp =
        (const v16h*)(pb + (((long)kb * nCoT + cot) << 9) + (lane << 4));
    v16h b = *bp;
    acc0 = __builtin_amdgcn_wmma_f32_16x16x32_f16(false, a0, false, b,
                                                  (short)0, acc0, false, false);
    acc1 = __builtin_amdgcn_wmma_f32_16x16x32_f16(false, a1, false, b,
                                                  (short)0, acc1, false, false);
  }

  if (valid) {
    int c = (cot << 4) + (lane & 15);
    if (c < Cout) {
      long obase = ((long)n * OH + oh) * OW;
#pragma unroll
      for (int r = 0; r < 8; ++r) {
        int mm = r + ((lane >> 4) << 3);         // C/D layout: M = r (+8 hi half)
        int ow = ow0 + mm;
        if (ow < OW)
          out[(obase + ow) * Ctot + coff + c] = (half_t)acc0[r];
        if (ow + 16 < OW)
          out[(obase + ow + 16) * Ctot + coff + c] = (half_t)acc1[r];
      }
    }
  }
}

// ------------------------------ WMMA GEMM ----------------------------------
// C[M,Nlog] = A[M,K] (f16 row-major, K % 32 == 0) * packedB. Output either
// f16 (outH) or f32 + bias (outF). Used for the 5x5->1x1 conv and the Linear.
__global__ __launch_bounds__(128) void
k_gemm_wmma(const half_t* __restrict__ A, const half_t* __restrict__ pb,
            int M, int K, int Nlog, const float* __restrict__ bias,
            float* __restrict__ outF, half_t* __restrict__ outH,
            int ldo, int nNT) {
  const int wid  = threadIdx.x >> 5;
  const int lane = threadIdx.x & 31;
  const int nMT = (M + 15) >> 4;
  long t = (long)blockIdx.x * WAVES_PER_WG + wid;
  if (t >= (long)nMT * nNT) return;
  const int nt = (int)(t % nNT);
  const int mt = (int)(t / nNT);

  const int row = (mt << 4) + (lane & 15);
  const bool rowOK = row < M;
  const int nKB = K >> 5;
  const half_t* arow = A + (long)row * K + ((lane & 16) ? 8 : 0);
  v8f acc = {};

  for (int kb = 0; kb < nKB; ++kb) {
    const half_t* ap = arow + (kb << 5);
    if (rowOK && kb + 1 < nKB)
      __builtin_prefetch(ap + 32, 0, 1);         // global_prefetch_b8 next K-step
    v8h lo = {}, hi = {};
    if (rowOK) { lo = *(const v8h*)ap; hi = *(const v8h*)(ap + 16); }
    v16h a;
#pragma unroll
    for (int e = 0; e < 8; ++e) { a[e] = lo[e]; a[e + 8] = hi[e]; }
    const v16h* bp =
        (const v16h*)(pb + (((long)kb * nNT + nt) << 9) + (lane << 4));
    v16h b = *bp;
    acc = __builtin_amdgcn_wmma_f32_16x16x32_f16(false, a, false, b,
                                                 (short)0, acc, false, false);
  }

  int c = (nt << 4) + (lane & 15);
  if (c < Nlog) {
#pragma unroll
    for (int r = 0; r < 8; ++r) {
      int mm = (mt << 4) + r + ((lane >> 4) << 3);
      if (mm < M) {
        if (outH) outH[(long)mm * ldo + c] = (half_t)acc[r];
        else      outF[(long)mm * ldo + c] = acc[r] + (bias ? bias[c] : 0.f);
      }
    }
  }
}

// ------------------------------ BatchNorm ----------------------------------
__global__ void k_bn_stats(const half_t* __restrict__ x, float* __restrict__ acc,
                           long total, int C) {
  __shared__ float ss[64], sq[64];
  for (int c = threadIdx.x; c < C; c += blockDim.x) { ss[c] = 0.f; sq[c] = 0.f; }
  __syncthreads();
  for (long i = (long)blockIdx.x * blockDim.x + threadIdx.x; i < total;
       i += (long)gridDim.x * blockDim.x) {
    float v = (float)x[i];
    int c = (int)(i % C);
    atomicAdd(&ss[c], v);          // ds_add_f32
    atomicAdd(&sq[c], v * v);
  }
  __syncthreads();
  for (int c = threadIdx.x; c < C; c += blockDim.x) {
    atomicAdd(&acc[c], ss[c]);     // global_atomic_add_f32
    atomicAdd(&acc[64 + c], sq[c]);
  }
}

__global__ void k_bn_apply(half_t* __restrict__ x, const float* __restrict__ acc,
                           const float* __restrict__ g, const float* __restrict__ b,
                           long total, int C, float invCnt) {
  long i = (long)blockIdx.x * blockDim.x + threadIdx.x;
  if (i >= total) return;
  int c = (int)(i % C);
  float mean = acc[c] * invCnt;
  float var  = acc[64 + c] * invCnt - mean * mean;
  float v = ((float)x[i] - mean) * rsqrtf(var + 1e-4f) * g[c] + b[c];
  x[i] = (half_t)fmaxf(v, 0.f);
}

// ------------------------------- MaxPool 3x3 s2 ----------------------------
__global__ void k_maxpool(const half_t* __restrict__ in, half_t* __restrict__ out,
                          int N, int IH, int IW, int C, int OH, int OW) {
  long i = (long)blockIdx.x * blockDim.x + threadIdx.x;
  long total = (long)N * OH * OW * C;
  if (i >= total) return;
  int c  = (int)(i % C);   long r = i / C;
  int ox = (int)(r % OW);  r /= OW;
  int oy = (int)(r % OH);
  int n  = (int)(r / OH);
  float m = -3.0e38f;
#pragma unroll
  for (int dy = 0; dy < 3; ++dy)
#pragma unroll
    for (int dx = 0; dx < 3; ++dx) {
      int iy = oy * 2 + dy, ix = ox * 2 + dx;
      m = fmaxf(m, (float)in[(((long)n * IH + iy) * IW + ix) * C + c]);
    }
  out[i] = (half_t)m;
}

// -------------------------- 3x3 s2 deconvolution ---------------------------
// out = (in-1)*2 + 3, VALID. Tiny (8->8 channels) -> direct VALU gather.
__global__ void k_deconv(const half_t* __restrict__ in, const float* __restrict__ w,
                         half_t* __restrict__ out,
                         int N, int IH, int IW, int Cb, int OH, int OW,
                         int Ctot, int coff) {
  long i = (long)blockIdx.x * blockDim.x + threadIdx.x;
  long total = (long)N * OH * OW * Cb;
  if (i >= total) return;
  int co = (int)(i % Cb);  long r = i / Cb;
  int ox = (int)(r % OW);  r /= OW;
  int oy = (int)(r % OH);
  int n  = (int)(r / OH);
  float s = 0.f;
#pragma unroll
  for (int ky = 0; ky < 3; ++ky) {
    int ty = oy - ky;
    if (ty < 0 || (ty & 1)) continue;
    int iy = ty >> 1;
    if (iy >= IH) continue;
#pragma unroll
    for (int kx = 0; kx < 3; ++kx) {
      int tx = ox - kx;
      if (tx < 0 || (tx & 1)) continue;
      int ix = tx >> 1;
      if (ix >= IW) continue;
      const half_t* ip = in + (((long)n * IH + iy) * IW + ix) * Cb;
      for (int ci = 0; ci < Cb; ++ci)
        s += (float)ip[ci] * w[((ci * Cb + co) * 3 + ky) * 3 + kx];
    }
  }
  out[(((long)n * OH + oy) * OW + ox) * Ctot + coff + co] = (half_t)s;
}

// ------------------------------- host driver -------------------------------
// d_in leaf order (JAX pytree: dict keys sorted alphabetically, lists in order)
// params: blocks[0..3]{bnd.b,bnd.g,bnj.b,bnj.g,bns.b,bns.g,wa,wd,ws,wu},
//         bn1.b,bn1.g,...,bnf.b,bnf.g, c1,c2,c3,c4,cf, lb, lw, then x.
enum {
  IB0 = 0, IB1 = 10, IB2 = 20, IB3 = 30,
  I_BN1B = 40, I_BN1G, I_BN2B, I_BN2G, I_BN3B, I_BN3G, I_BN4B, I_BN4G,
  I_BNFB, I_BNFG,
  I_C1 = 50, I_C2, I_C3, I_C4, I_CF, I_LB, I_LW, I_X
};

extern "C" void kernel_launch(void* const* d_in, const int* in_sizes, int n_in,
                              void* d_out, int out_size, void* d_ws, size_t ws_size,
                              hipStream_t stream) {
  (void)in_sizes; (void)n_in; (void)out_size; (void)ws_size;
  const int NB = 512;

  size_t cur = 0;
  auto allocH = [&](size_t halves) -> half_t* {
    cur = (cur + 511) & ~(size_t)511;
    half_t* p = (half_t*)((char*)d_ws + cur);
    cur += halves * sizeof(half_t);
    return p;
  };

  // Arenas (f16 halves), sequentially reused through the pipeline (~180 MB).
  half_t* BUF0 = allocH(37000000);          // up to 512*95*95*8
  half_t* BUF1 = allocH(37000000);
  half_t* BUF2 = allocH(9600000);           // xh / p1 / z chain / final f16
  half_t* BUF3 = allocH(4700000);           // p2 / p3 / p4
  float*  STATS = (float*)allocH(256);      // 64 sums + 64 sumsq
  half_t* Z  = BUF2;
  half_t* ZS = BUF2 + 600000;
  half_t* F5 = BUF2;                        // 512 x 64 after final conv

  auto packW = [&](int idx, int Cout, int Cin, int KH, int KW) -> half_t* {
    int Ktot = Cin * KH * KW;
    long tot = (long)((Ktot + 31) >> 5) * ((Cout + 15) >> 4) * 512;
    half_t* p = allocH((size_t)tot);
    k_pack<<<dim3((unsigned)((tot + 255) / 256)), dim3(256), 0, stream>>>(
        (const float*)d_in[idx], p, Cout, Cin, KH, KW);
    return p;
  };
  auto conv = [&](const half_t* in, const half_t* pb, half_t* out,
                  int H, int W, int Cin, int Cout, int stride, int pad,
                  int OH, int OW, int Ctot, int coff) {
    long tiles = (long)NB * OH * ((OW + 31) >> 5) * ((Cout + 15) >> 4);
    k_conv_wmma<<<dim3((unsigned)((tiles + WAVES_PER_WG - 1) / WAVES_PER_WG)),
                  dim3(32 * WAVES_PER_WG), 0, stream>>>(
        in, pb, out, NB, H, W, Cin, Cout, 3, 3, stride, pad, OH, OW, Ctot, coff);
  };
  auto bn = [&](half_t* x, long total, int C, int gIdx, int bIdx) {
    k_zero<<<dim3(1), dim3(128), 0, stream>>>(STATS, 128);
    k_bn_stats<<<dim3(1024), dim3(256), 0, stream>>>(x, STATS, total, C);
    k_bn_apply<<<dim3((unsigned)((total + 255) / 256)), dim3(256), 0, stream>>>(
        x, STATS, (const float*)d_in[gIdx], (const float*)d_in[bIdx],
        total, C, (float)C / (float)total);
  };
  auto pool = [&](const half_t* in, half_t* out, int IH, int IW, int C,
                  int OH, int OW) {
    long total = (long)NB * OH * OW * C;
    k_maxpool<<<dim3((unsigned)((total + 255) / 256)), dim3(256), 0, stream>>>(
        in, out, NB, IH, IW, C, OH, OW);
  };
  auto runBlock = [&](half_t* in, int H, int nin, int a, int b, int bi,
                      half_t* join) {
    int Hh = (H - 3) / 2 + 1;
    half_t* pwa = packW(bi + 6, a, nin, 3, 3);
    conv(in, pwa, join, H, H, nin, a, 1, 1, H, H, a + b, 0);      // ya
    half_t* pwd = packW(bi + 7, b, nin, 3, 3);
    conv(in, pwd, Z, H, H, nin, b, 2, 0, Hh, Hh, b, 0);           // s2 conv
    bn(Z, (long)NB * Hh * Hh * b, b, bi + 1, bi + 0);             // bnd
    half_t* pws = packW(bi + 8, b, b, 3, 3);
    conv(Z, pws, ZS, Hh, Hh, b, b, 1, 1, Hh, Hh, b, 0);           // submanifold
    bn(ZS, (long)NB * Hh * Hh * b, b, bi + 5, bi + 4);            // bns
    long dt = (long)NB * H * H * b;
    k_deconv<<<dim3((unsigned)((dt + 255) / 256)), dim3(256), 0, stream>>>(
        ZS, (const float*)d_in[bi + 9], join, NB, Hh, Hh, b, H, H, a + b, a);
    bn(join, (long)NB * H * H * (a + b), a + b, bi + 3, bi + 2);  // bnj
  };

  // ---- pipeline ----
  half_t* XH = BUF2;
  long nx = (long)NB * 95 * 95;
  k_f32_to_f16<<<dim3((unsigned)((nx + 255) / 256)), dim3(256), 0, stream>>>(
      (const float*)d_in[I_X], XH, nx);

  half_t* p;
  p = packW(I_C1, 8, 1, 3, 3);
  conv(XH, p, BUF0, 95, 95, 1, 8, 1, 1, 95, 95, 8, 0);
  bn(BUF0, (long)NB * 95 * 95 * 8, 8, I_BN1G, I_BN1B);
  p = packW(I_C2, 8, 8, 3, 3);
  conv(BUF0, p, BUF1, 95, 95, 8, 8, 1, 1, 95, 95, 8, 0);
  bn(BUF1, (long)NB * 95 * 95 * 8, 8, I_BN2G, I_BN2B);
  pool(BUF1, BUF2, 95, 95, 8, 47, 47);                 // xh dead -> p1 in BUF2
  p = packW(I_C3, 16, 8, 3, 3);
  conv(BUF2, p, BUF0, 47, 47, 8, 16, 1, 1, 47, 47, 16, 0);
  bn(BUF0, (long)NB * 47 * 47 * 16, 16, I_BN3G, I_BN3B);
  p = packW(I_C4, 16, 16, 3, 3);
  conv(BUF0, p, BUF1, 47, 47, 16, 16, 1, 1, 47, 47, 16, 0);
  bn(BUF1, (long)NB * 47 * 47 * 16, 16, I_BN4G, I_BN4B);
  pool(BUF1, BUF3, 47, 47, 16, 23, 23);                // p2

  runBlock(BUF3, 23, 16, 16, 8, IB0, BUF0);            // 16 -> 24
  runBlock(BUF0, 23, 24, 16, 8, IB1, BUF1);            // 24 -> 24
  pool(BUF1, BUF3, 23, 23, 24, 11, 11);                // p3
  runBlock(BUF3, 11, 24, 24, 8, IB2, BUF0);            // 24 -> 32
  runBlock(BUF0, 11, 32, 24, 8, IB3, BUF1);            // 32 -> 32
  pool(BUF1, BUF3, 11, 11, 32, 5, 5);                  // p4: 512 x (5*5*32=800)

  // Final 5x5 VALID conv == GEMM: A = p4 rows of 800 halves, K = 800 = 25*32.
  p = packW(I_CF, 64, 32, 5, 5);
  {
    long tiles = (long)((512 + 15) / 16) * ((64 + 15) / 16);
    k_gemm_wmma<<<dim3((unsigned)((tiles + 3) / 4)), dim3(128), 0, stream>>>(
        BUF3, p, 512, 800, 64, nullptr, nullptr, F5, 64, (64 + 15) / 16);
  }
  bn(F5, (long)512 * 64, 64, I_BNFG, I_BNFB);

  // Linear(64 -> 100) + bias, f32 output.
  p = packW(I_LW, 100, 64, 1, 1);
  {
    long tiles = (long)((512 + 15) / 16) * ((100 + 15) / 16);
    k_gemm_wmma<<<dim3((unsigned)((tiles + 3) / 4)), dim3(128), 0, stream>>>(
        F5, p, 512, 64, 100, (const float*)d_in[I_LB], (float*)d_out, nullptr,
        100, (100 + 15) / 16);
  }
}
